// BiLSTMCRF_27831388078552
// MI455X (gfx1250) — matrix operand order repaired and checked by reading, not compiled, not run on hardware
//
#include <hip/hip_runtime.h>

// BiLSTM-CRF forward for MI455X (gfx1250, wave32, WMMA bf16 16x16x32, f32 accum).
// Workspace use ~82 MB. All buffers fully (re)initialized each call.

#define B_    32
#define S_    256
#define EMB_  256
#define HID_  256
#define NLAB_ 9
#define G4H   1024   // 4*HID

typedef __attribute__((ext_vector_type(16))) __bf16       v16bf;
typedef __attribute__((ext_vector_type(8)))  float        v8f;
typedef __attribute__((ext_vector_type(4)))  unsigned int u32x4;

union Frag16 { u32x4 q[2]; v16bf v; };

__device__ __forceinline__ unsigned short f2bf(float f) {
  unsigned int u = __float_as_uint(f);
  u += 0x7fffu + ((u >> 16) & 1u);           // round-to-nearest-even
  return (unsigned short)(u >> 16);
}
__device__ __forceinline__ float sigf(float x) { return 1.f / (1.f + __expf(-x)); }
__device__ __forceinline__ float tanhfast(float x) {
  x = fminf(fmaxf(x, -15.f), 15.f);
  float e = __expf(2.f * x);
  return (e - 1.f) / (e + 1.f);
}

// ---------------------------------------------------------------------------
// Repack W [Nrows x K] (row-major f32, used as B = W^T) into WMMA B-fragment
// order: frag[(nt*kT + kt)*32 + lane][j] with N = nt*16 + (lane&15),
// K = kt*32 + (lane>>4)*16 + j. Rows >= Nrows are zero-padded.
__global__ void prep_wfrag(const float* __restrict__ W, unsigned short* __restrict__ F,
                           int Nrows, int K, int kT) {
  int bt   = blockIdx.x;           // nt*kT + kt
  int kt   = bt % kT;
  int nt   = bt / kT;
  int lane = threadIdx.x;
  int n    = nt * 16 + (lane & 15);
  int kb   = kt * 32 + (lane >> 4) * 16;
  unsigned short* out = F + ((size_t)bt * 32 + lane) * 16;
#pragma unroll
  for (int j = 0; j < 16; ++j) {
    float v = (n < Nrows) ? W[(size_t)n * K + kb + j] : 0.f;
    out[j] = f2bf(v);
  }
}

__global__ void prep_bias(const float* __restrict__ a, const float* __restrict__ b,
                          float* __restrict__ out, int n, int nvalid) {
  int i = blockIdx.x * blockDim.x + threadIdx.x;
  if (i >= n) return;
  float v = 0.f;
  if (i < nvalid) { v = a[i]; if (b) v += b[i]; }
  out[i] = v;
}

// x_bf16[row][e] = bf16(emb[ids[row]][e]), row = b*S + s
__global__ void embed_bf16(const int* __restrict__ ids, const float* __restrict__ emb,
                           unsigned short* __restrict__ x) {
  int row = blockIdx.x;
  int id  = ids[row];
  x[(size_t)row * EMB_ + threadIdx.x] = f2bf(emb[(size_t)id * EMB_ + threadIdx.x]);
}

// ---------------------------------------------------------------------------
// Y[M x Nld] = X[M x K](bf16) @ W^T (fragment-packed bf16) + bias, f32 accum.
// One wave per 16x16 output tile; 8 waves per block.
__global__ void gemm_bf16(const unsigned short* __restrict__ X,
                          const unsigned short* __restrict__ Wfrag,
                          const float* __restrict__ bias,
                          float* __restrict__ Y,
                          int M, int K, int kT, int nT, int Nld) {
  int wave = threadIdx.x >> 5;
  int lane = threadIdx.x & 31;
  int tile = blockIdx.x * (blockDim.x >> 5) + wave;
  int rowTiles = M >> 4;
  if (tile >= rowTiles * nT) return;
  int mt = tile / nT, nt = tile % nT;
  int hi = lane >> 4, nl = lane & 15;

  const unsigned short* xrow = X + (size_t)(mt * 16 + nl) * K;
  v8f acc = {};
  for (int kt = 0; kt < kT; ++kt) {
    Frag16 a, b;
    const unsigned short* ap = xrow + kt * 32 + hi * 16;    // 16B-aligned chunks
    a.q[0] = *(const u32x4*)(ap);
    a.q[1] = *(const u32x4*)(ap + 8);
    const unsigned short* bp = Wfrag + (((size_t)nt * kT + kt) * 32 + lane) * 16;
    b.q[0] = *(const u32x4*)(bp);
    b.q[1] = *(const u32x4*)(bp + 8);
    acc = __builtin_amdgcn_wmma_f32_16x16x32_bf16(false, a.v, false, b.v,
                                                  (short)0, acc, false, false);
  }
  int n = nt * 16 + nl;
  float bv = bias ? bias[n] : 0.f;
#pragma unroll
  for (int r = 0; r < 8; ++r)
    Y[(size_t)(mt * 16 + r + 8 * hi) * Nld + n] = acc[r] + bv;
}

// ---------------------------------------------------------------------------
// Sequential LSTM scan, one workgroup per direction (blockIdx.x: 0=fwd, 1=rev).
// Per step: g = xg[t] + h @ Whh^T via WMMA (h bf16 in LDS, Whh fragments from
// L2), then elementwise gate update of c/h in LDS. 1024 threads = 32 waves;
// wave w owns m-tile (w&1) and 4 of the 64 n-tiles. LDS: g 128K + c 32K + h 16K.
__global__ void lstm_scan(const float* __restrict__ xgF, const float* __restrict__ xgR,
                          const unsigned short* __restrict__ whhF,
                          const unsigned short* __restrict__ whhR,
                          unsigned short* __restrict__ hcat) {
  extern __shared__ char smem[];
  float*          g_s = (float*)smem;                         // [32][1024]
  float*          c_s = g_s + B_ * G4H;                       // [32][256]
  unsigned short* h_s = (unsigned short*)(c_s + B_ * HID_);   // [32][256]

  const int reverse = blockIdx.x;
  const float* xg = reverse ? xgR : xgF;
  const unsigned short* whh = reverse ? whhR : whhF;
  const int colOff = reverse ? HID_ : 0;

  int tid = threadIdx.x;
  for (int i = tid; i < B_ * HID_; i += 1024) { c_s[i] = 0.f; h_s[i] = 0; }
  __syncthreads();

  int wave = tid >> 5, lane = tid & 31;
  int hi = lane >> 4, nl = lane & 15;
  int mt = wave & 1, ng = wave >> 1;                 // ng: 0..15, 4 n-tiles each

  for (int step = 0; step < S_; ++step) {
    int t = reverse ? (S_ - 1 - step) : step;

    v8f acc[4] = {{}, {}, {}, {}};
    const unsigned short* hrow = h_s + (mt * 16 + nl) * HID_;
    for (int kt = 0; kt < 8; ++kt) {
      Frag16 a;
      const unsigned short* ap = hrow + kt * 32 + hi * 16;   // ds_load_b128 x2
      a.q[0] = *(const u32x4*)(ap);
      a.q[1] = *(const u32x4*)(ap + 8);
      const unsigned short* bp =
          whh + (((size_t)(ng * 4) * 8 + kt) * 32 + lane) * 16;
#pragma unroll
      for (int q = 0; q < 4; ++q) {
        Frag16 b;
        b.q[0] = *(const u32x4*)(bp);
        b.q[1] = *(const u32x4*)(bp + 8);
        acc[q] = __builtin_amdgcn_wmma_f32_16x16x32_bf16(false, a.v, false, b.v,
                                                         (short)0, acc[q], false, false);
        bp += 8 * 32 * 16;                                   // next n-tile
      }
    }
#pragma unroll
    for (int q = 0; q < 4; ++q) {
      int n = (ng * 4 + q) * 16 + nl;
#pragma unroll
      for (int r = 0; r < 8; ++r)
        g_s[(mt * 16 + r + 8 * hi) * G4H + n] = acc[q][r];
    }
    __syncthreads();

    for (int idx = tid; idx < B_ * HID_; idx += 1024) {
      int m = idx >> 8, hh = idx & 255;
      size_t xb = ((size_t)(m * S_ + t)) * G4H + hh;
      const float* gr = g_s + m * G4H + hh;
      float gi = gr[0]        + xg[xb];
      float gf = gr[HID_]     + xg[xb + HID_];
      float gn = gr[2 * HID_] + xg[xb + 2 * HID_];
      float go = gr[3 * HID_] + xg[xb + 3 * HID_];
      float c  = sigf(gf) * c_s[idx] + sigf(gi) * tanhfast(gn);
      c_s[idx] = c;
      unsigned short hb = f2bf(sigf(go) * tanhfast(c));
      h_s[idx] = hb;
      hcat[((size_t)(m * S_ + t)) * (2 * HID_) + colOff + hh] = hb;
    }
    __syncthreads();
  }
}

// ---------------------------------------------------------------------------
// CRF: gold score + forward algorithm + mean(fwd - gold). Single block.
// emissions stored [B*S][16] (cols 9..15 are padding, never read).
__global__ void crf_loss(const float* __restrict__ emis,
                         const int* __restrict__ labels,
                         const float* __restrict__ mask,
                         const float* __restrict__ trans,
                         float* __restrict__ out) {
  __shared__ float tr[NLAB_ * NLAB_];
  __shared__ float alpha[2][B_][NLAB_];
  __shared__ float gold[B_];
  __shared__ float res[B_];
  int tid = threadIdx.x;
  if (tid < NLAB_ * NLAB_) tr[tid] = trans[tid];
  __syncthreads();

  if (tid < B_) {                                   // gold path score
    int b = tid;
    int prev = labels[b * S_];
    float g = emis[(size_t)(b * S_) * 16 + prev];
    for (int t = 1; t < S_; ++t) {
      int l = labels[b * S_ + t];
      g += (emis[(size_t)(b * S_ + t) * 16 + l] + tr[prev * NLAB_ + l]) * mask[b * S_ + t];
      prev = l;
    }
    gold[b] = g;
  }

  int b = tid / NLAB_, j = tid % NLAB_;
  bool act = tid < B_ * NLAB_;
  if (act) alpha[0][b][j] = emis[(size_t)(b * S_) * 16 + j];
  __syncthreads();

  for (int t = 1; t < S_; ++t) {                    // forward recursion
    int cur = t & 1, prv = cur ^ 1;
    if (act) {
      float m = -3.4e38f;
#pragma unroll
      for (int i = 0; i < NLAB_; ++i) m = fmaxf(m, alpha[prv][b][i] + tr[i * NLAB_ + j]);
      float s = 0.f;
#pragma unroll
      for (int i = 0; i < NLAB_; ++i) s += __expf(alpha[prv][b][i] + tr[i * NLAB_ + j] - m);
      alpha[cur][b][j] = emis[(size_t)(b * S_ + t) * 16 + j] + m + __logf(s);
    }
    __syncthreads();
  }

  if (tid < B_) {
    int fin = (S_ - 1) & 1;
    float m = -3.4e38f;
    for (int i = 0; i < NLAB_; ++i) m = fmaxf(m, alpha[fin][tid][i]);
    float s = 0.f;
    for (int i = 0; i < NLAB_; ++i) s += __expf(alpha[fin][tid][i] - m);
    res[tid] = (m + __logf(s)) - gold[tid];
  }
  __syncthreads();
  if (tid == 0) {
    float s = 0.f;
    for (int i = 0; i < B_; ++i) s += res[i];
    out[0] = s / (float)B_;
  }
}

// ---------------------------------------------------------------------------
extern "C" void kernel_launch(void* const* d_in, const int* in_sizes, int n_in,
                              void* d_out, int out_size, void* d_ws, size_t ws_size,
                              hipStream_t stream) {
  (void)in_sizes; (void)n_in; (void)out_size; (void)ws_size;
  const int*   ids     = (const int*)  d_in[0];
  const int*   labels  = (const int*)  d_in[1];
  const float* mask    = (const float*)d_in[2];
  const float* emb     = (const float*)d_in[3];
  const float* W_out   = (const float*)d_in[4];
  const float* b_out   = (const float*)d_in[5];
  const float* trans   = (const float*)d_in[6];
  const float* Wih_l0  = (const float*)d_in[7];
  const float* Whh_l0  = (const float*)d_in[8];
  const float* bih_l0  = (const float*)d_in[9];
  const float* bhh_l0  = (const float*)d_in[10];
  const float* Wih_l0r = (const float*)d_in[11];
  const float* Whh_l0r = (const float*)d_in[12];
  const float* bih_l0r = (const float*)d_in[13];
  const float* bhh_l0r = (const float*)d_in[14];
  const float* Wih_l1  = (const float*)d_in[15];
  const float* Whh_l1  = (const float*)d_in[16];
  const float* bih_l1  = (const float*)d_in[17];
  const float* bhh_l1  = (const float*)d_in[18];
  const float* Wih_l1r = (const float*)d_in[19];
  const float* Whh_l1r = (const float*)d_in[20];
  const float* bih_l1r = (const float*)d_in[21];
  const float* bhh_l1r = (const float*)d_in[22];

  char* base = (char*)d_ws;
  size_t off = 0;
  auto take = [&](size_t bytes) -> void* {
    void* r = base + off;
    off += (bytes + 255) & ~(size_t)255;
    return r;
  };

  unsigned short* x_bf  = (unsigned short*)take((size_t)B_ * S_ * EMB_ * 2);      // 4 MB
  float*          xg_a  = (float*)take((size_t)B_ * S_ * G4H * 4);                // 32 MB
  float*          xg_b  = (float*)take((size_t)B_ * S_ * G4H * 4);                // 32 MB
  unsigned short* hcat  = (unsigned short*)take((size_t)B_ * S_ * 2 * HID_ * 2);  // 8 MB
  unsigned short* fWihL0f = (unsigned short*)take((size_t)64 * 8  * 512 * 2);
  unsigned short* fWihL0r = (unsigned short*)take((size_t)64 * 8  * 512 * 2);
  unsigned short* fWhhL0f = (unsigned short*)take((size_t)64 * 8  * 512 * 2);
  unsigned short* fWhhL0r = (unsigned short*)take((size_t)64 * 8  * 512 * 2);
  unsigned short* fWihL1f = (unsigned short*)take((size_t)64 * 16 * 512 * 2);
  unsigned short* fWihL1r = (unsigned short*)take((size_t)64 * 16 * 512 * 2);
  unsigned short* fWhhL1f = (unsigned short*)take((size_t)64 * 8  * 512 * 2);
  unsigned short* fWhhL1r = (unsigned short*)take((size_t)64 * 8  * 512 * 2);
  unsigned short* fWout   = (unsigned short*)take((size_t)1  * 16 * 512 * 2);
  float* biasL0f = (float*)take(G4H * 4);
  float* biasL0r = (float*)take(G4H * 4);
  float* biasL1f = (float*)take(G4H * 4);
  float* biasL1r = (float*)take(G4H * 4);
  float* bout16  = (float*)take(16 * 4);
  float* emis    = (float*)take((size_t)B_ * S_ * 16 * 4);                        // 512 KB

  // 1) Weight/bias repack (bf16, WMMA fragment order)
  prep_wfrag<<<64 * 8,  32, 0, stream>>>(Wih_l0,  fWihL0f, 1024, 256, 8);
  prep_wfrag<<<64 * 8,  32, 0, stream>>>(Wih_l0r, fWihL0r, 1024, 256, 8);
  prep_wfrag<<<64 * 8,  32, 0, stream>>>(Whh_l0,  fWhhL0f, 1024, 256, 8);
  prep_wfrag<<<64 * 8,  32, 0, stream>>>(Whh_l0r, fWhhL0r, 1024, 256, 8);
  prep_wfrag<<<64 * 16, 32, 0, stream>>>(Wih_l1,  fWihL1f, 1024, 512, 16);
  prep_wfrag<<<64 * 16, 32, 0, stream>>>(Wih_l1r, fWihL1r, 1024, 512, 16);
  prep_wfrag<<<64 * 8,  32, 0, stream>>>(Whh_l1,  fWhhL1f, 1024, 256, 8);
  prep_wfrag<<<64 * 8,  32, 0, stream>>>(Whh_l1r, fWhhL1r, 1024, 256, 8);
  prep_wfrag<<<1 * 16,  32, 0, stream>>>(W_out,   fWout,   NLAB_, 512, 16);
  prep_bias<<<4, 256, 0, stream>>>(bih_l0,  bhh_l0,  biasL0f, G4H, G4H);
  prep_bias<<<4, 256, 0, stream>>>(bih_l0r, bhh_l0r, biasL0r, G4H, G4H);
  prep_bias<<<4, 256, 0, stream>>>(bih_l1,  bhh_l1,  biasL1f, G4H, G4H);
  prep_bias<<<4, 256, 0, stream>>>(bih_l1r, bhh_l1r, biasL1r, G4H, G4H);
  prep_bias<<<1, 256, 0, stream>>>(b_out, (const float*)nullptr, bout16, 16, NLAB_);

  // 2) Embedding gather (f32 -> bf16)
  embed_bf16<<<B_ * S_, 256, 0, stream>>>(ids, emb, x_bf);

  // 3) Layer 0: input-gate GEMMs (M=8192,K=256,N=1024), then bidirectional scan
  gemm_bf16<<<4096, 256, 0, stream>>>(x_bf, fWihL0f, biasL0f, xg_a, B_ * S_, 256, 8, 64, G4H);
  gemm_bf16<<<4096, 256, 0, stream>>>(x_bf, fWihL0r, biasL0r, xg_b, B_ * S_, 256, 8, 64, G4H);
  const size_t scanLds = (size_t)B_ * G4H * 4 + (size_t)B_ * HID_ * 4 + (size_t)B_ * HID_ * 2;
  lstm_scan<<<2, 1024, scanLds, stream>>>(xg_a, xg_b, fWhhL0f, fWhhL0r, hcat);

  // 4) Layer 1: input-gate GEMMs from hcat (K=512), then scan (overwrites hcat)
  gemm_bf16<<<4096, 256, 0, stream>>>(hcat, fWihL1f, biasL1f, xg_a, B_ * S_, 512, 16, 64, G4H);
  gemm_bf16<<<4096, 256, 0, stream>>>(hcat, fWihL1r, biasL1r, xg_b, B_ * S_, 512, 16, 64, G4H);
  lstm_scan<<<2, 1024, scanLds, stream>>>(xg_a, xg_b, fWhhL1f, fWhhL1r, hcat);

  // 5) Emissions via WMMA (N padded 9->16), then CRF loss -> scalar
  gemm_bf16<<<64, 256, 0, stream>>>(hcat, fWout, bout16, emis, B_ * S_, 512, 16, 1, 16);
  crf_loss<<<1, 320, 0, stream>>>(emis, labels, mask, trans, (float*)d_out);
}
